// PixelContrastiveLoss_68083821576868
// MI455X (gfx1250) — matrix operand order repaired and checked by reading, not compiled, not run on hardware
//
#include <hip/hip_runtime.h>
#include <hip/hip_bf16.h>

typedef __attribute__((ext_vector_type(16))) _Float16 v16h;
typedef __attribute__((ext_vector_type(8)))  _Float16 half8;
typedef __attribute__((ext_vector_type(8)))  float    v8f;

#define NPIX   8192      // B*H*W = 2*64*64
#define CCH    128       // channels
#define HWSZ   4096      // H*W
#define RTILES (NPIX/16) // 512 row tiles (target pixels)
#define CBLKS  (NPIX/32) // 256 column blocks (32 input pixels each)
#define WAVES  8         // 256 threads / wave32

// ---------------------------------------------------------------------------
// Kernel 1: per-pixel inverse L2 norm over channels + seg labels -> bytes
// grid: (NPIX/256, 2)  y==0 -> input, y==1 -> target
// ---------------------------------------------------------------------------
__global__ void k_norm_seg(const float* __restrict__ inp, const float* __restrict__ tgt,
                           const int* __restrict__ iseg, const int* __restrict__ tseg,
                           float* __restrict__ inv_i, float* __restrict__ inv_t,
                           unsigned char* __restrict__ pi8, unsigned char* __restrict__ pt8) {
    const int n  = blockIdx.x * 256 + threadIdx.x;          // pixel id (b*HW + hw)
    const float* src = blockIdx.y ? tgt : inp;
    const int b  = n >> 12;            // n / HWSZ
    const int hw = n & (HWSZ - 1);
    const size_t base = (size_t)b * (size_t)CCH * HWSZ + (size_t)hw;
    float ss = 0.f;
#pragma unroll 4
    for (int c = 0; c < CCH; ++c) {
        float v = src[base + (size_t)c * HWSZ];
        ss += v * v;
    }
    const float inv = 1.0f / fmaxf(sqrtf(ss), 1e-12f);
    if (blockIdx.y) { inv_t[n] = inv; pt8[n] = (unsigned char)tseg[n]; }
    else            { inv_i[n] = inv; pi8[n] = (unsigned char)iseg[n]; }
}

// ---------------------------------------------------------------------------
// Kernel 2: normalize + convert to row-major (NPIX x CCH) f16
// grid: (NPIX*CCH/256, 2)
// ---------------------------------------------------------------------------
__global__ void k_convert(const float* __restrict__ inp, const float* __restrict__ tgt,
                          const float* __restrict__ inv_i, const float* __restrict__ inv_t,
                          _Float16* __restrict__ ni, _Float16* __restrict__ nt) {
    const unsigned g = blockIdx.x * 256 + threadIdx.x;
    const unsigned c = g >> 13;          // g / NPIX
    const unsigned n = g & (NPIX - 1);   // pixel fastest -> coalesced reads
    const float* src; const float* invn; _Float16* dst;
    if (blockIdx.y) { src = tgt; invn = inv_t; dst = nt; }
    else            { src = inp; invn = inv_i; dst = ni; }
    const int b  = n >> 12;
    const int hw = n & (HWSZ - 1);
    const float v = src[(size_t)b * (size_t)CCH * HWSZ + (size_t)c * HWSZ + hw] * invn[n];
    dst[(size_t)n * CCH + c] = (_Float16)v;
}

// ---------------------------------------------------------------------------
// Kernel 3: contrastive accumulation, double-buffered, scalar loop control.
// One block per 32-column slab (two 16-col WMMA tiles; B fragments pinned in
// registers). The block's 8 waves split the 512 row tiles of target pixels;
// the wave id is readfirstlane'd into an SGPR so loop control is fully
// scalar (no exec-mask predication around the WMMA body). Per row tile:
// 8x v_wmma_f32_16x16x32_f16 (K=128, 2 col tiles), raw v_exp_f32 + masked
// accumulate. The NEXT tile's A fragments + seg bytes are prefetched with a
// branchless wrapped index, so each iteration's s_wait_loadcnt covers loads
// issued a full iteration earlier.
//
// Per-lane WMMA layouts (wave32, CDNA5 ISA 7.12.2):
//   A (16x32 f16): row M = lane&15; halves i<8 -> K = 32*kb + koffA + i,
//                  i>=8 -> K = 32*kb + 16 + koffA + (i-8), koffA = (lane>>4)*8
//   B (32x16 f16): col N = lane&15; same K order as A (i<8 at 32*kb + koffA + i, i>=8 at 32*kb + 16 + koffA + (i-8))
//   D (16x16 f32): VGPR r -> M = r + 8*(lane>>4), N = lane&15
// ---------------------------------------------------------------------------
__global__ void __launch_bounds__(256)
k_contrast(const _Float16* __restrict__ nt, const _Float16* __restrict__ ni,
           const unsigned char* __restrict__ pt8, const unsigned char* __restrict__ pi8,
           float* __restrict__ partial) {
    __shared__ float s_n0[WAVES * 32];
    __shared__ float s_d0[WAVES * 32];
    __shared__ float s_n1[WAVES * 32];
    __shared__ float s_d1[WAVES * 32];

    const int tid   = threadIdx.x;
    const int lane  = tid & 31;
    // Scalar (SGPR) wave id: loop control + tile indices stay uniform/scalar.
    const int wave  = __builtin_amdgcn_readfirstlane(tid) >> 5;
    const int col   = lane & 15;
    const int hi16  = lane >> 4;          // 0 or 1
    const int koffA = hi16 << 3;          // 0 or 8
    const int koffB = hi16 << 4;          // 0 or 16
    const int n0    = blockIdx.x * 32;    // first input-pixel column of slab

    // Preload B fragments for both column tiles; pinned in registers.
    v16h bf0[4], bf1[4];
    {
        const _Float16* __restrict__ br0 = ni + (size_t)(n0 + col) * CCH;
        const _Float16* __restrict__ br1 = ni + (size_t)(n0 + 16 + col) * CCH;
#pragma unroll
        for (int kb = 0; kb < 4; ++kb) {
            half8 lo = *(const half8*)(br0 + kb * 32 + koffA);
            half8 hi = *(const half8*)(br0 + kb * 32 + koffA + 16);
            bf0[kb] = __builtin_shufflevector(lo, hi, 0,1,2,3,4,5,6,7,8,9,10,11,12,13,14,15);
            lo = *(const half8*)(br1 + kb * 32 + koffA);
            hi = *(const half8*)(br1 + kb * 32 + koffA + 16);
            bf1[kb] = __builtin_shufflevector(lo, hi, 0,1,2,3,4,5,6,7,8,9,10,11,12,13,14,15);
        }
    }
    const unsigned pib0 = pi8[n0 + col];
    const unsigned pib1 = pi8[n0 + 16 + col];
    constexpr float SCALE = (1.0f / 0.07f) * 1.44269504088896340736f; // /tau, base-2

    float an0 = 0.f, ad0 = 0.f, an1 = 0.f, ad1 = 0.f;

    // ---- prologue: load first tile's A fragments + seg bytes ----
    int t = wave;
    v16h af[4];
    uint2 pv;
    {
        const _Float16* __restrict__ arow = nt + (size_t)(t * 16 + col) * CCH + koffA;
        pv = *(const uint2*)(pt8 + t * 16 + koffA);
#pragma unroll
        for (int kb = 0; kb < 4; ++kb) {
            const half8 lo = *(const half8*)(arow + kb * 32);
            const half8 hi = *(const half8*)(arow + kb * 32 + 16);
            af[kb] = __builtin_shufflevector(lo, hi, 0,1,2,3,4,5,6,7,8,9,10,11,12,13,14,15);
        }
    }

    for (int it = 0; it < RTILES / WAVES; ++it) {
        // ---- prefetch next tile; wrapped index => branchless, always valid ----
        const int tn = (t + WAVES) & (RTILES - 1);
        v16h afn[4];
        uint2 pvn;
        {
            const _Float16* __restrict__ arow = nt + (size_t)(tn * 16 + col) * CCH + koffA;
            pvn = *(const uint2*)(pt8 + tn * 16 + koffA);
#pragma unroll
            for (int kb = 0; kb < 4; ++kb) {
                const half8 lo = *(const half8*)(arow + kb * 32);
                const half8 hi = *(const half8*)(arow + kb * 32 + 16);
                afn[kb] = __builtin_shufflevector(lo, hi, 0,1,2,3,4,5,6,7,8,9,10,11,12,13,14,15);
            }
        }

        // ---- compute on current tile ----
        v8f acc0 = {};
        v8f acc1 = {};
#pragma unroll
        for (int kb = 0; kb < 4; ++kb) {
            acc0 = __builtin_amdgcn_wmma_f32_16x16x32_f16(false, af[kb], false, bf0[kb],
                                                          (short)0, acc0, false, false);
            acc1 = __builtin_amdgcn_wmma_f32_16x16x32_f16(false, af[kb], false, bf1[kb],
                                                          (short)0, acc1, false, false);
        }
#pragma unroll
        for (int r = 0; r < 8; ++r) {
            const unsigned sb = ((r < 4) ? (pv.x >> (8 * r)) : (pv.y >> (8 * (r - 4)))) & 0xffu;
            const float e0 = __builtin_amdgcn_exp2f(acc0[r] * SCALE); // raw v_exp_f32
            const float e1 = __builtin_amdgcn_exp2f(acc1[r] * SCALE);
            ad0 += e0;
            ad1 += e1;
            an0 += (sb == pib0) ? e0 : 0.f;
            an1 += (sb == pib1) ? e1 : 0.f;
        }

        t = tn;
#pragma unroll
        for (int kb = 0; kb < 4; ++kb) af[kb] = afn[kb];
        pv = pvn;
    }

    s_n0[wave * 32 + lane] = an0;
    s_d0[wave * 32 + lane] = ad0;
    s_n1[wave * 32 + lane] = an1;
    s_d1[wave * 32 + lane] = ad1;
    __syncthreads();

    if (tid < 32) {   // one thread per slab column: deterministic-order reduction
        const int c = tid & 15;
        const float* sn = (tid < 16) ? s_n0 : s_n1;
        const float* sd = (tid < 16) ? s_d0 : s_d1;
        float num = 0.f, den = 0.f;
#pragma unroll
        for (int w = 0; w < WAVES; ++w) {
            num += sn[w * 32 + c] + sn[w * 32 + c + 16];
            den += sd[w * 32 + c] + sd[w * 32 + c + 16];
        }
        const float term = -logf(num / (den + 1e-8f)) * (1.0f / (float)NPIX);
        s_n0[tid] = term;
    }
    __syncthreads();
    if (tid == 0) {
        float s = 0.f;
        for (int i = 0; i < 32; ++i) s += s_n0[i];
        partial[blockIdx.x] = s;
    }
}

// ---------------------------------------------------------------------------
// Kernel 4: final deterministic reduction of 256 block partials -> loss
// ---------------------------------------------------------------------------
__global__ void k_reduce(const float* __restrict__ partial, float* __restrict__ out) {
    __shared__ float s[CBLKS];
    const int tid = threadIdx.x;
    s[tid] = partial[tid];
    __syncthreads();
    for (int off = CBLKS / 2; off > 0; off >>= 1) {
        if (tid < off) s[tid] += s[tid + off];
        __syncthreads();
    }
    if (tid == 0) out[0] = s[0];
}

// ---------------------------------------------------------------------------
extern "C" void kernel_launch(void* const* d_in, const int* in_sizes, int n_in,
                              void* d_out, int out_size, void* d_ws, size_t ws_size,
                              hipStream_t stream) {
    (void)in_sizes; (void)n_in; (void)out_size; (void)ws_size;
    const float* inp  = (const float*)d_in[0];
    const float* tgt  = (const float*)d_in[1];
    const int*   iseg = (const int*)d_in[2];
    const int*   tseg = (const int*)d_in[3];
    float* out = (float*)d_out;

    char* ws = (char*)d_ws;
    _Float16* ni = (_Float16*)ws;                                   // 2 MB
    _Float16* nt = (_Float16*)(ws + (size_t)NPIX * CCH * 2);        // 2 MB
    float* inv_i = (float*)(ws + (size_t)4 * 1024 * 1024);          // 32 KB
    float* inv_t = inv_i + NPIX;                                    // 32 KB
    unsigned char* pi8 = (unsigned char*)(inv_t + NPIX);            // 8 KB
    unsigned char* pt8 = pi8 + NPIX;                                // 8 KB
    float* partial = (float*)(pt8 + NPIX);                          // 1 KB

    dim3 g1(NPIX / 256, 2);
    k_norm_seg<<<g1, 256, 0, stream>>>(inp, tgt, iseg, tseg, inv_i, inv_t, pi8, pt8);

    dim3 g2((NPIX * CCH) / 256, 2);
    k_convert<<<g2, 256, 0, stream>>>(inp, tgt, inv_i, inv_t, ni, nt);

    k_contrast<<<CBLKS, 256, 0, stream>>>(nt, ni, pt8, pi8, partial);

    k_reduce<<<1, CBLKS, 0, stream>>>(partial, out);
}
